// MoE_27041114095775
// MI455X (gfx1250) — matrix operand order repaired and checked by reading, not compiled, not run on hardware
//
#include <hip/hip_runtime.h>
#include <cstdint>
#include <cstddef>

#define T_TOK 2048
#define HDIM  2048
#define NEXP  16
#define IDIM  768
#define TOPK  4
#define MT    4                 // m-tiles per wave (register blocking)
#define MROWS (MT * 16)         // 64 rows per wave-group

typedef __attribute__((ext_vector_type(16))) __bf16 bf16x16;
typedef __attribute__((ext_vector_type(8)))  float  f32x8;

union Frag16 { bf16x16 v; int4 d[2]; };

// A fragment (bf16, row-major). 16-bit A 16x32 layout: lanes 0-15: row M=l,
// K-chunks [k0,k0+8) and [k0+16,k0+24); lanes 16-31: chunks +8.
__device__ inline bf16x16 load_a_bf16(const __bf16* __restrict__ base, int ld,
                                      int row, int k0, int hi) {
    Frag16 a;
    const __bf16* p = base + (size_t)row * ld + k0 + hi * 8;
    a.d[0] = *(const int4*)(p);
    a.d[1] = *(const int4*)(p + 16);
    return a.v;
}

__device__ inline bf16x16 load_a_ptr(const __bf16* __restrict__ rowp, int k0, int hi) {
    Frag16 a;
    const __bf16* p = rowp + k0 + hi * 8;
    a.d[0] = *(const int4*)(p);
    a.d[1] = *(const int4*)(p + 16);
    return a.v;
}

// B fragment (bf16 weights, row-major [N,K]): lane holds column N=l%16,
// K-chunks [k0,k0+8) and [k0+16,k0+24); lanes 16-31: chunks +8. Two b128 loads.
__device__ inline bf16x16 load_b_bf16(const __bf16* __restrict__ base, int ld,
                                      int row, int k0, int hi) {
    Frag16 b;
    const __bf16* p = base + (size_t)row * ld + k0 + hi * 8;
    b.d[0] = *(const int4*)(p);
    b.d[1] = *(const int4*)(p + 16);
    return b.v;
}

#define WMMA_BF16(a, b, c) \
    __builtin_amdgcn_wmma_f32_16x16x32_bf16(false, (a), false, (b), (short)0, (c), false, false)

// ---------------- fp32 -> bf16 convert ----------------
__global__ void __launch_bounds__(256)
cvt_bf16_kernel(const float* __restrict__ x, __bf16* __restrict__ xb, int n) {
    int i = (blockIdx.x * blockDim.x + threadIdx.x) * 4;
    if (i >= n) return;
    float4 f = *(const float4*)(x + i);
    xb[i + 0] = (__bf16)f.x; xb[i + 1] = (__bf16)f.y;
    xb[i + 2] = (__bf16)f.z; xb[i + 3] = (__bf16)f.w;
}

// ---------------- router: sigmoid scores ----------------
__global__ void __launch_bounds__(256)
router_scores_kernel(const float* __restrict__ x, const float* __restrict__ rw,
                     float* __restrict__ scores) {
    int tid = blockIdx.x * blockDim.x + threadIdx.x;
    if (tid >= T_TOK * NEXP) return;
    int t = tid / NEXP, e = tid % NEXP;
    const float* xr = x + (size_t)t * HDIM;
    const float* wr = rw + (size_t)e * HDIM;
    float acc = 0.f;
    for (int k = 0; k < HDIM; k += 4) {
        float4 a = *(const float4*)(xr + k);
        float4 b = *(const float4*)(wr + k);
        acc += a.x * b.x + a.y * b.y + a.z * b.z + a.w * b.w;
    }
    scores[tid] = __builtin_amdgcn_rcpf(1.f + __expf(-acc));
}

// ---------------- router: top-4 + normalize ----------------
__global__ void __launch_bounds__(128)
router_topk_kernel(const float* __restrict__ scores, const float* __restrict__ ebias,
                   float* __restrict__ combine, int* __restrict__ topk_idx) {
    int t = blockIdx.x * blockDim.x + threadIdx.x;
    if (t >= T_TOK) return;
    float s[NEXP], sb[NEXP];
#pragma unroll
    for (int e = 0; e < NEXP; ++e) {
        s[e]  = scores[t * NEXP + e];
        sb[e] = s[e] + ebias[e];
    }
    int idx[TOPK];
    float wsum = 0.f;
#pragma unroll
    for (int k = 0; k < TOPK; ++k) {
        int best = 0; float bv = -1e30f;
#pragma unroll
        for (int e = 0; e < NEXP; ++e) {
            bool taken = false;
#pragma unroll
            for (int j = 0; j < TOPK; ++j) taken |= (j < k) && (idx[j] == e);
            if (!taken && sb[e] > bv) { bv = sb[e]; best = e; }
        }
        idx[k] = best;
        wsum += s[best];
    }
    float inv = 1.f / (wsum + 1e-20f);   // ROUTED_SCALE == 1.0
#pragma unroll
    for (int e = 0; e < NEXP; ++e) combine[t * NEXP + e] = 0.f;
#pragma unroll
    for (int k = 0; k < TOPK; ++k) {
        combine[t * NEXP + idx[k]] = s[idx[k]] * inv;
        topk_idx[t * TOPK + k] = idx[k];
    }
}

// ---------------- deterministic per-expert token lists (no atomics) ----------------
__global__ void __launch_bounds__(32)
build_lists_kernel(const int* __restrict__ topk_idx, int* __restrict__ lists,
                   int* __restrict__ counts) {
    int e = threadIdx.x;
    if (e >= NEXP) return;
    int cnt = 0;
    int* lst = lists + (size_t)e * T_TOK;
    for (int t = 0; t < T_TOK; ++t) {
        int4 id = *(const int4*)(topk_idx + t * TOPK);
        if (id.x == e || id.y == e || id.z == e || id.w == e) lst[cnt++] = t;
    }
    counts[e] = cnt;
}

// ---------------- dense pass 1 (shared expert) ----------------
__global__ void __launch_bounds__(256)
moe_up_swiglu_kernel(const __bf16* __restrict__ xb, const __bf16* __restrict__ wg,
                     const __bf16* __restrict__ wu, __bf16* __restrict__ act) {
    const int lane = threadIdx.x, wave = threadIdx.y;
    const int job  = blockIdx.x * 8 + wave;
    const int nti  = IDIM / 16;
    const int mg   = job / nti;
    const int nt   = job % nti;
    const int hi   = lane >> 4, rl = lane & 15;
    const int brow = nt * 16 + rl;

    f32x8 cg[MT] = {}, cu[MT] = {};
    for (int k0 = 0; k0 < HDIM; k0 += 32) {
        bf16x16 bg = load_b_bf16(wg, HDIM, brow, k0, hi);
        bf16x16 bu = load_b_bf16(wu, HDIM, brow, k0, hi);
#pragma unroll
        for (int j = 0; j < MT; ++j) {
            bf16x16 a = load_a_bf16(xb, HDIM, mg * MROWS + j * 16 + rl, k0, hi);
            cg[j] = WMMA_BF16(a, bg, cg[j]);
            cu[j] = WMMA_BF16(a, bu, cu[j]);
        }
    }
    const int col = nt * 16 + rl;
#pragma unroll
    for (int j = 0; j < MT; ++j) {
        int rbase = mg * MROWS + j * 16 + hi * 8;
#pragma unroll
        for (int v = 0; v < 8; ++v) {
            float g = cg[j][v], u = cu[j][v];
            float sig = __builtin_amdgcn_rcpf(1.f + __expf(-g));
            act[(size_t)(rbase + v) * IDIM + col] = (__bf16)(g * sig * u);
        }
    }
}

// ---------------- dense pass 2 (shared expert, write mode) ----------------
__global__ void __launch_bounds__(256)
moe_down_dense_kernel(const __bf16* __restrict__ act, const __bf16* __restrict__ wd,
                      float* __restrict__ out) {
    const int lane = threadIdx.x, wave = threadIdx.y;
    const int job  = blockIdx.x * 8 + wave;
    const int nth  = HDIM / 16;
    const int mg   = job / nth;
    const int nt   = job % nth;
    const int hi   = lane >> 4, rl = lane & 15;
    const int brow = nt * 16 + rl;

    f32x8 c[MT] = {};
    for (int k0 = 0; k0 < IDIM; k0 += 32) {
        bf16x16 b = load_b_bf16(wd, IDIM, brow, k0, hi);
#pragma unroll
        for (int j = 0; j < MT; ++j) {
            bf16x16 a = load_a_bf16(act, IDIM, mg * MROWS + j * 16 + rl, k0, hi);
            c[j] = WMMA_BF16(a, b, c[j]);
        }
    }
    const int col = nt * 16 + rl;
#pragma unroll
    for (int j = 0; j < MT; ++j) {
        int rbase = mg * MROWS + j * 16 + hi * 8;
#pragma unroll
        for (int v = 0; v < 8; ++v)
            out[(size_t)(rbase + v) * HDIM + col] = c[j][v];
    }
}

// ---------------- sparse pass 1: gathered rows -> compact act ----------------
__global__ void __launch_bounds__(256)
moe_up_sparse_kernel(const __bf16* __restrict__ xb, const __bf16* __restrict__ wg,
                     const __bf16* __restrict__ wu, __bf16* __restrict__ act,
                     const int* __restrict__ list, const int* __restrict__ count_ptr) {
    const int lane = threadIdx.x, wave = threadIdx.y;
    const int job  = blockIdx.x * 8 + wave;
    const int nti  = IDIM / 16;
    const int mg   = job / nti;
    const int nt   = job % nti;
    const int cnt  = *count_ptr;            // wave-uniform
    if (mg * MROWS >= cnt) return;          // uniform wave exit, EXEC stays full
    const int hi = lane >> 4, rl = lane & 15;
    const int brow = nt * 16 + rl;

    // K-invariant row gather: one base pointer per m-tile per lane.
    const __bf16* aptr[MT];
#pragma unroll
    for (int j = 0; j < MT; ++j) {
        int r = mg * MROWS + j * 16 + rl;
        int tok = list[r < cnt ? r : cnt - 1];
        aptr[j] = xb + (size_t)tok * HDIM;
    }

    f32x8 cg[MT] = {}, cu[MT] = {};
    for (int k0 = 0; k0 < HDIM; k0 += 32) {
        bf16x16 bg = load_b_bf16(wg, HDIM, brow, k0, hi);
        bf16x16 bu = load_b_bf16(wu, HDIM, brow, k0, hi);
#pragma unroll
        for (int j = 0; j < MT; ++j) {
            bf16x16 a = load_a_ptr(aptr[j], k0, hi);
            cg[j] = WMMA_BF16(a, bg, cg[j]);
            cu[j] = WMMA_BF16(a, bu, cu[j]);
        }
    }
    const int col = nt * 16 + rl;
#pragma unroll
    for (int j = 0; j < MT; ++j) {
        int rbase = mg * MROWS + j * 16 + hi * 8;
#pragma unroll
        for (int v = 0; v < 8; ++v) {
            int r = rbase + v;
            if (r < cnt) {
                float g = cg[j][v], u = cu[j][v];
                float sig = __builtin_amdgcn_rcpf(1.f + __expf(-g));
                act[(size_t)r * IDIM + col] = (__bf16)(g * sig * u);
            }
        }
    }
}

// ---------------- sparse pass 2: compact act -> scatter-accumulate out ----------------
__global__ void __launch_bounds__(256)
moe_down_sparse_kernel(const __bf16* __restrict__ act, const __bf16* __restrict__ wd,
                       const float* __restrict__ combine, int expert,
                       const int* __restrict__ list, const int* __restrict__ count_ptr,
                       float* __restrict__ out) {
    const int lane = threadIdx.x, wave = threadIdx.y;
    const int job  = blockIdx.x * 8 + wave;
    const int nth  = HDIM / 16;
    const int mg   = job / nth;
    const int nt   = job % nth;
    const int cnt  = *count_ptr;
    if (mg * MROWS >= cnt) return;
    const int hi = lane >> 4, rl = lane & 15;
    const int brow = nt * 16 + rl;

    f32x8 c[MT] = {};
    for (int k0 = 0; k0 < IDIM; k0 += 32) {
        bf16x16 b = load_b_bf16(wd, IDIM, brow, k0, hi);
#pragma unroll
        for (int j = 0; j < MT; ++j) {
            bf16x16 a = load_a_bf16(act, IDIM, mg * MROWS + j * 16 + rl, k0, hi);
            c[j] = WMMA_BF16(a, b, c[j]);
        }
    }
    const int col = nt * 16 + rl;
#pragma unroll
    for (int j = 0; j < MT; ++j) {
        int rbase = mg * MROWS + j * 16 + hi * 8;
#pragma unroll
        for (int v = 0; v < 8; ++v) {
            int r = rbase + v;
            if (r < cnt) {
                int tok = list[r];
                float scale = combine[tok * NEXP + expert];
                size_t idx = (size_t)tok * HDIM + col;
                out[idx] += scale * c[j][v];   // unique (tok,col) per launch
            }
        }
    }
}

extern "C" void kernel_launch(void* const* d_in, const int* in_sizes, int n_in,
                              void* d_out, int out_size, void* d_ws, size_t ws_size,
                              hipStream_t stream) {
    const float* x   = (const float*)d_in[0];   // [T, H]
    const float* rw  = (const float*)d_in[1];   // [E, H]
    const float* eb  = (const float*)d_in[2];   // [E]
    const float* gup = (const float*)d_in[3];   // [E, 2I, H]
    const float* dwn = (const float*)d_in[4];   // [E, H, I]
    const float* sg  = (const float*)d_in[5];   // [IS, H]
    const float* su  = (const float*)d_in[6];   // [IS, H]
    const float* sd  = (const float*)d_in[7];   // [H, IS]
    float* out = (float*)d_out;

    char* ws = (char*)d_ws;
    size_t off = 0;
    __bf16* xb   = (__bf16*)(ws + off); off += (size_t)T_TOK * HDIM * 2;     // 8 MB
    __bf16* act  = (__bf16*)(ws + off); off += (size_t)T_TOK * IDIM * 2;     // 3 MB
    __bf16* wgub = (__bf16*)(ws + off); off += (size_t)2 * IDIM * HDIM * 2;  // 6 MB
    __bf16* wdb  = (__bf16*)(ws + off); off += (size_t)HDIM * IDIM * 2;      // 3 MB
    float* scores   = (float*)(ws + off); off += (size_t)T_TOK * NEXP * 4;
    float* combine  = (float*)(ws + off); off += (size_t)T_TOK * NEXP * 4;
    int*   topk_idx = (int*)(ws + off);   off += (size_t)T_TOK * TOPK * 4;
    int*   lists    = (int*)(ws + off);   off += (size_t)NEXP * T_TOK * 4;
    int*   counts   = (int*)(ws + off);   off += 64;

    // Router + dispatch metadata (deterministic, atomics-free).
    cvt_bf16_kernel<<<(T_TOK * HDIM / 4 + 255) / 256, 256, 0, stream>>>(x, xb, T_TOK * HDIM);
    router_scores_kernel<<<(T_TOK * NEXP + 255) / 256, 256, 0, stream>>>(x, rw, scores);
    router_topk_kernel<<<(T_TOK + 127) / 128, 128, 0, stream>>>(scores, eb, combine, topk_idx);
    build_lists_kernel<<<1, 32, 0, stream>>>(topk_idx, lists, counts);

    dim3 blk(32, 8);
    const int g1 = (T_TOK / MROWS) * (IDIM / 16) / 8;   // 192 blocks (worst case)
    const int g2 = (T_TOK / MROWS) * (HDIM / 16) / 8;   // 512 blocks (worst case)
    const int nGU = 2 * IDIM * HDIM;
    const int nD  = HDIM * IDIM;

    // Shared expert (dense over all tokens); pass-2 initializes out.
    cvt_bf16_kernel<<<(nD / 4 + 255) / 256, 256, 0, stream>>>(sg, wgub, nD);
    cvt_bf16_kernel<<<(nD / 4 + 255) / 256, 256, 0, stream>>>(su, wgub + (size_t)IDIM * HDIM, nD);
    cvt_bf16_kernel<<<(nD / 4 + 255) / 256, 256, 0, stream>>>(sd, wdb, nD);
    moe_up_swiglu_kernel<<<g1, blk, 0, stream>>>(xb, wgub, wgub + (size_t)IDIM * HDIM, act);
    moe_down_dense_kernel<<<g2, blk, 0, stream>>>(act, wdb, out);

    // Routed experts: sparse gather/scatter, ~4x fewer FLOPs than dense.
    for (int e = 0; e < NEXP; ++e) {
        const float* wg32 = gup + (size_t)e * 2 * IDIM * HDIM;
        const float* wd32 = dwn + (size_t)e * HDIM * IDIM;
        cvt_bf16_kernel<<<(nGU / 4 + 255) / 256, 256, 0, stream>>>(wg32, wgub, nGU);
        cvt_bf16_kernel<<<(nD / 4 + 255) / 256, 256, 0, stream>>>(wd32, wdb, nD);
        moe_up_sparse_kernel<<<g1, blk, 0, stream>>>(xb, wgub, wgub + (size_t)IDIM * HDIM,
                                                     act, lists + (size_t)e * T_TOK, counts + e);
        moe_down_sparse_kernel<<<g2, blk, 0, stream>>>(act, wdb, combine, e,
                                                       lists + (size_t)e * T_TOK, counts + e, out);
    }
}
